// VectorizedSlidingWindowModel_3693671874779
// MI455X (gfx1250) — matrix operand order repaired and checked by reading, not compiled, not run on hardware
//
#include <hip/hip_runtime.h>

#define S_LEN  4096
#define DMODEL 1024
#define NHEAD  16
#define HD     64
#define WIN    64
#define BSTRIDE 136   // padded LDS row stride (elems): 272B = 68 dwords -> bank spread

typedef __bf16 bf16_t;
typedef __attribute__((ext_vector_type(16))) __bf16 v16bf;
typedef __attribute__((ext_vector_type(8)))  __bf16 v8bf;
typedef __attribute__((ext_vector_type(4)))  __bf16 v4bf;
typedef __attribute__((ext_vector_type(8)))  float  v8f;
typedef __attribute__((ext_vector_type(4)))  float  v4f;

#define WMMA_BF16(a, b, c) \
  __builtin_amdgcn_wmma_f32_16x16x32_bf16(false, (a), false, (b), (short)0, (c), false, false)

// A-fragment: lane holds one M-row; K packed as {0..7,16..23} (half 0) / {8..15,24..31} (half 1).
__device__ __forceinline__ v16bf frag_a_load(const bf16_t* p) {
  v8bf lo = *(const v8bf*)(p);
  v8bf hi = *(const v8bf*)(p + 16);
  return __builtin_shufflevector(lo, hi, 0,1,2,3,4,5,6,7,8,9,10,11,12,13,14,15);
}

// B-fragment: lane holds one N-column; K packed as {0..7,16..23} (half 0) / {8..15,24..31} (half 1), row stride ld.
__device__ __forceinline__ v16bf frag_b_load(const bf16_t* p, int ld) {
  v16bf r;
#pragma unroll
  for (int s = 0; s < 8; ++s) { r[s] = p[(size_t)s * ld]; r[s + 8] = p[(size_t)(16 + s) * ld]; }
  return r;
}

// 16B global -> LDS, asynchronous (ASYNCcnt-tracked, no VGPR staging).
__device__ __forceinline__ void async_cp16(const bf16_t* g, bf16_t* l) {
  unsigned loff = (unsigned)(uintptr_t)l;                 // low 32 bits = LDS offset
  unsigned long long ga = (unsigned long long)(uintptr_t)g;
  asm volatile("global_load_async_to_lds_b128 %0, %1, off"
               :: "v"(loff), "v"(ga) : "memory");
}
__device__ __forceinline__ void wait_async0() {
  asm volatile("s_wait_asynccnt 0x0" ::: "memory");
}

// ---------------- f32 -> bf16 convert (vectorized x4) ----------------
__global__ void cvt_f32_bf16(const float* __restrict__ src, bf16_t* __restrict__ dst, int n) {
  int i = (blockIdx.x * blockDim.x + threadIdx.x) * 4;
  if (i + 3 < n) {
    v4f v = *(const v4f*)(src + i);
    v4bf o;
    o[0] = (bf16_t)v[0]; o[1] = (bf16_t)v[1]; o[2] = (bf16_t)v[2]; o[3] = (bf16_t)v[3];
    *(v4bf*)(dst + i) = o;
  }
}

// -------- weight transpose + convert: src[n][k] f32 -> dst[k][n] bf16 --------
__global__ void cvt_transpose_bf16(const float* __restrict__ src, bf16_t* __restrict__ dst,
                                   int Nrows, int Kcols) {
  int k = blockIdx.x * 32 + (threadIdx.x & 31);
  int n = blockIdx.y * 8 + (threadIdx.x >> 5);
  if (k < Kcols && n < Nrows)
    dst[(size_t)k * Nrows + n] = (bf16_t)src[(size_t)n * Kcols + k];
}

// ---------------- WMMA GEMM: Y[M][N] = A[M][K] * Wt[K][N] ----------------
// B tile (shared by all 8 waves) is double-buffered in LDS via async loads.
// store modes: 0 = bf16 [M][N]; 1 = bf16 [H][M][64] (Q/V); 2 = bf16 [H][64][M] (K^T); 3 = f32 [M][N]
__global__ __launch_bounds__(256)
void gemm_wmma_bf16(const bf16_t* __restrict__ A, const bf16_t* __restrict__ Wt,
                    void* __restrict__ out, int M, int N, int K, int mode) {
  __shared__ __align__(16) bf16_t bufB[2][32 * BSTRIDE];

  const int tid  = threadIdx.x;
  const int lane = tid & 31;
  const int wv   = tid >> 5;                   // 8 waves
  const int half = lane >> 4;
  const int nl   = lane & 15;
  const int wm   = wv & 3, wn = wv >> 2;       // 4x2 wave grid -> 128x128 block tile
  const int rowBase = blockIdx.x * 128 + wm * 32;
  const int colBlk  = blockIdx.y * 128;
  const int colBase = colBlk + wn * 64;

  // staging map: thread -> rows {tid>>4, tid>>4 + 16}, 8-elem col chunk
  const int srow = tid >> 4;                   // 0..15
  const int scol = (tid & 15) * 8;             // 0..120

  v8f acc[2][4];
#pragma unroll
  for (int mi = 0; mi < 2; ++mi)
#pragma unroll
    for (int ni = 0; ni < 4; ++ni)
      acc[mi][ni] = (v8f){0.f,0.f,0.f,0.f,0.f,0.f,0.f,0.f};

  // prologue: stage k0 = 0 into buffer 0
#pragma unroll
  for (int cc = 0; cc < 2; ++cc) {
    const int r = srow + cc * 16;
    async_cp16(Wt + (size_t)r * N + colBlk + scol, &bufB[0][r * BSTRIDE + scol]);
  }

  const int nsteps = K >> 5;
  for (int i = 0; i < nsteps; ++i) {
    const int k0 = i << 5;
    const int st = i & 1;

    wait_async0();                 // my async writes into bufB[st] have landed
    __syncthreads();               // everyone's writes landed; prev reads of bufB[st^1] done

    if (i + 1 < nsteps) {          // overlap next tile's async loads with this tile's WMMAs
#pragma unroll
      for (int cc = 0; cc < 2; ++cc) {
        const int r = srow + cc * 16;
        async_cp16(Wt + (size_t)(k0 + 32 + r) * N + colBlk + scol,
                   &bufB[st ^ 1][r * BSTRIDE + scol]);
      }
    }

    v16bf a[2], b[4];
#pragma unroll
    for (int mi = 0; mi < 2; ++mi)
      a[mi] = frag_a_load(A + (size_t)(rowBase + mi * 16 + nl) * K + k0 + half * 8);
#pragma unroll
    for (int ni = 0; ni < 4; ++ni)
      b[ni] = frag_b_load(&bufB[st][(half * 8) * BSTRIDE + wn * 64 + ni * 16 + nl], BSTRIDE);

#pragma unroll
    for (int mi = 0; mi < 2; ++mi)
#pragma unroll
      for (int ni = 0; ni < 4; ++ni)
        acc[mi][ni] = WMMA_BF16(a[mi], b[ni], acc[mi][ni]);
  }

#pragma unroll
  for (int mi = 0; mi < 2; ++mi) {
#pragma unroll
    for (int ni = 0; ni < 4; ++ni) {
      const int n  = colBase + ni * 16 + nl;
      const int m0 = rowBase + mi * 16 + half * 8;
      if (mode == 0) {
        bf16_t* o = (bf16_t*)out;
#pragma unroll
        for (int r = 0; r < 8; ++r) o[(size_t)(m0 + r) * N + n] = (bf16_t)acc[mi][ni][r];
      } else if (mode == 1) {
        bf16_t* o = (bf16_t*)out;
        const int h = n >> 6, d = n & 63;
#pragma unroll
        for (int r = 0; r < 8; ++r) o[((size_t)h * M + (m0 + r)) * HD + d] = (bf16_t)acc[mi][ni][r];
      } else if (mode == 2) {
        bf16_t* o = (bf16_t*)out;
        const int h = n >> 6, d = n & 63;
#pragma unroll
        for (int r = 0; r < 8; ++r) o[((size_t)h * HD + d) * M + (m0 + r)] = (bf16_t)acc[mi][ni][r];
      } else {
        float* o = (float*)out;
#pragma unroll
        for (int r = 0; r < 8; ++r) o[(size_t)(m0 + r) * N + n] = acc[mi][ni][r];
      }
    }
  }
}

// ---------------- sliding-window attention, 1 wave per (head, 16-query block) ----------------
// Q: [H][S][64] bf16, Kt: [H][64][S] bf16, V: [H][S][64] bf16, O: [S][1024] bf16
__global__ __launch_bounds__(256)
void swa_wmma(const bf16_t* __restrict__ Q, const bf16_t* __restrict__ Kt,
              const bf16_t* __restrict__ V, bf16_t* __restrict__ O) {
  const int lane = threadIdx.x & 31;
  const int w    = threadIdx.x >> 5;
  const int half = lane >> 4;
  const int nl   = lane & 15;
  const int wave = blockIdx.x * 8 + w;
  const int h    = wave >> 8;                  // S/16 = 256 query blocks
  const int qb   = wave & 255;

  __shared__ __align__(16) bf16_t plds[8][16 * 96];

  const bf16_t* Qh = Q  + (size_t)h * S_LEN * HD;
  const bf16_t* Kh = Kt + (size_t)h * HD * S_LEN;
  const bf16_t* Vh = V  + (size_t)h * S_LEN * HD;

  // Q fragments (A): rows = 16 queries, K = hd in two steps of 32
  const int qrow = qb * 16 + nl;
  v16bf qa0 = frag_a_load(Qh + (size_t)qrow * HD +      half * 8);
  v16bf qa1 = frag_a_load(Qh + (size_t)qrow * HD + 32 + half * 8);

  // 96-key superset [16*qb-80, 16*qb+15] of the 64-wide causal window
  const int key0 = qb * 16 - 80;

  // ---- scores: 6 tiles of 16 keys, each = 2 chained WMMAs over hd ----
  v8f sc[6];
#pragma unroll
  for (int t = 0; t < 6; ++t) {
    int kbase = key0 + t * 16;              // 16-aligned: whole tile >=0 or <0
    int kclmp = kbase < 0 ? 0 : kbase;
    v16bf b0 = frag_b_load(Kh + (size_t)(half * 8)      * S_LEN + kclmp + nl, S_LEN);
    v16bf b1 = frag_b_load(Kh + (size_t)(32 + half * 8) * S_LEN + kclmp + nl, S_LEN);
    v8f c = (v8f){0.f,0.f,0.f,0.f,0.f,0.f,0.f,0.f};
    c = WMMA_BF16(qa0, b0, c);
    c = WMMA_BF16(qa1, b1, c);
    const int key = kbase + nl;             // this lane's score column
#pragma unroll
    for (int r = 0; r < 8; ++r) {
      const int qabs = qb * 16 + r + half * 8;
      const bool valid = (key >= 0) && (key <= qabs) && (key > qabs - WIN);
      c[r] = valid ? c[r] * 0.125f : -__builtin_inff();   // 1/sqrt(64)
    }
    sc[t] = c;
  }

  // ---- masked softmax over 96 columns per row (16-lane shfl tree within a half) ----
  float inv[8];
#pragma unroll
  for (int r = 0; r < 8; ++r) {
    float m = sc[0][r];
#pragma unroll
    for (int t = 1; t < 6; ++t) m = fmaxf(m, sc[t][r]);
#pragma unroll
    for (int s = 1; s < 16; s <<= 1) m = fmaxf(m, __shfl_xor(m, s, 32));
    float sum = 0.f;
#pragma unroll
    for (int t = 0; t < 6; ++t) { float e = __expf(sc[t][r] - m); sc[t][r] = e; sum += e; }
#pragma unroll
    for (int s = 1; s < 16; s <<= 1) sum += __shfl_xor(sum, s, 32);
    inv[r] = 1.f / sum;
  }

  // ---- re-lay P (D layout) into LDS row-major [16 q][96 key] for A-frag reads ----
#pragma unroll
  for (int t = 0; t < 6; ++t)
#pragma unroll
    for (int r = 0; r < 8; ++r)
      plds[w][(r + half * 8) * 96 + t * 16 + nl] = (bf16_t)(sc[t][r] * inv[r]);

  // ---- O = P @ V : 3 K-tiles of 32 keys x 4 hd-subtiles ----
  v8f o[4];
#pragma unroll
  for (int ds = 0; ds < 4; ++ds) o[ds] = (v8f){0.f,0.f,0.f,0.f,0.f,0.f,0.f,0.f};

#pragma unroll
  for (int kt = 0; kt < 3; ++kt) {
    v16bf pa = frag_a_load(&plds[w][nl * 96 + kt * 32 + half * 8]);
    const int key = key0 + kt * 32 + half * 8;      // B-frag element = key row of V
#pragma unroll
    for (int ds = 0; ds < 4; ++ds) {
      v16bf vb;
#pragma unroll
      for (int s = 0; s < 8; ++s) {
        const int k1 = key + s, k2 = key + 16 + s;  // masked keys have p == 0
        vb[s]     = Vh[(size_t)(k1 < 0 ? 0 : k1) * HD + ds * 16 + nl];
        vb[s + 8] = Vh[(size_t)(k2 < 0 ? 0 : k2) * HD + ds * 16 + nl];
      }
      o[ds] = WMMA_BF16(pa, vb, o[ds]);
    }
  }

  // ---- store attention output as [s][D_MODEL] bf16 for the final projection ----
#pragma unroll
  for (int ds = 0; ds < 4; ++ds) {
    const int d = ds * 16 + nl;
#pragma unroll
    for (int r = 0; r < 8; ++r) {
      const int srow = qb * 16 + r + half * 8;
      O[(size_t)srow * DMODEL + h * HD + d] = (bf16_t)o[ds][r];
    }
  }
}

extern "C" void kernel_launch(void* const* d_in, const int* in_sizes, int n_in,
                              void* d_out, int out_size, void* d_ws, size_t ws_size,
                              hipStream_t stream) {
  (void)in_sizes; (void)n_in; (void)out_size; (void)ws_size;
  const float* x  = (const float*)d_in[0];
  const float* Wq = (const float*)d_in[1];
  const float* Wk = (const float*)d_in[2];
  const float* Wv = (const float*)d_in[3];
  const float* Wo = (const float*)d_in[4];
  float* out = (float*)d_out;

  const size_t MB = 1u << 20;
  char* ws = (char*)d_ws;
  bf16_t* xbf   = (bf16_t*)(ws);             // 8 MB  [S][D]
  bf16_t* wq_t  = (bf16_t*)(ws +  8 * MB);   // 2 MB  [K][N]
  bf16_t* wk_t  = (bf16_t*)(ws + 10 * MB);
  bf16_t* wv_t  = (bf16_t*)(ws + 12 * MB);
  bf16_t* wo_t  = (bf16_t*)(ws + 14 * MB);
  bf16_t* qbuf  = (bf16_t*)(ws + 16 * MB);   // 8 MB  [H][S][64]
  bf16_t* ktbuf = (bf16_t*)(ws + 24 * MB);   // 8 MB  [H][64][S]
  bf16_t* vbuf  = (bf16_t*)(ws + 32 * MB);   // 8 MB  [H][S][64]
  bf16_t* abuf  = (bf16_t*)(ws + 40 * MB);   // 8 MB  [S][D]

  const int NX = S_LEN * DMODEL;             // 4,194,304
  cvt_f32_bf16<<<NX / 1024, 256, 0, stream>>>(x, xbf, NX);

  dim3 tg(DMODEL / 32, DMODEL / 8);
  cvt_transpose_bf16<<<tg, 256, 0, stream>>>(Wq, wq_t, DMODEL, DMODEL);
  cvt_transpose_bf16<<<tg, 256, 0, stream>>>(Wk, wk_t, DMODEL, DMODEL);
  cvt_transpose_bf16<<<tg, 256, 0, stream>>>(Wv, wv_t, DMODEL, DMODEL);
  cvt_transpose_bf16<<<tg, 256, 0, stream>>>(Wo, wo_t, DMODEL, DMODEL);

  dim3 gg(S_LEN / 128, DMODEL / 128);
  gemm_wmma_bf16<<<gg, 256, 0, stream>>>(xbf, wq_t, qbuf,  S_LEN, DMODEL, DMODEL, 1);
  gemm_wmma_bf16<<<gg, 256, 0, stream>>>(xbf, wk_t, ktbuf, S_LEN, DMODEL, DMODEL, 2);
  gemm_wmma_bf16<<<gg, 256, 0, stream>>>(xbf, wv_t, vbuf,  S_LEN, DMODEL, DMODEL, 1);

  swa_wmma<<<(NHEAD * (S_LEN / 16)) / 8, 256, 0, stream>>>(qbuf, ktbuf, vbuf, abuf);

  gemm_wmma_bf16<<<gg, 256, 0, stream>>>(abuf, wo_t, out, S_LEN, DMODEL, DMODEL, 3);
}
